// MonoSDF_84834194030725
// MI455X (gfx1250) — matrix operand order repaired and checked by reading, not compiled, not run on hardware
//
#include <hip/hip_runtime.h>
#include <hip/hip_bf16.h>
#include <math.h>

typedef __attribute__((ext_vector_type(16))) _Float16 v16h;
typedef __attribute__((ext_vector_type(8)))  _Float16 v8h;
typedef __attribute__((ext_vector_type(8)))  float    v8f;

#define HID 256

// ---------------- workspace layout (bytes) ----------------
// All weight arrays are stored PRE-SWIZZLED into WMMA B-fragment layout:
//   [kk][nt][lane(32)][j(16)] f16, 1KB per 32x16 tile, so a wave loads a
//   whole B fragment with two coalesced global b128 loads.
#define OFF_GW1   0u          // 1 kk * 16 nt  (K-pad 11->32)
#define OFF_GW2   16384u      // 8 kk * 16 nt
#define OFF_GW2T  147456u     // 8 kk * 16 nt  (W2^T)
#define OFF_GW3   278528u     // 8 kk * 1 nt   (N-pad 9->16)
#define OFF_GW1T  286720u     // 8 kk * 1 nt   (W1^T, N-pad 11->16)
#define OFF_CW1   294912u     // 1 kk * 16 nt  (K-pad 14->32)
#define OFF_CW2   311296u     // 8 kk * 16 nt
#define OFF_CW3   442368u     // 8 kk * 1 nt   (N-pad 3->16)
#define OFF_SDF   450560u     // f32 [262144]
#define OFF_RGB   1499136u    // f32 [262144*3]

// ---------------- WMMA helpers ----------------
__device__ __forceinline__ v8f wmma16(v16h a, v16h b, v8f c) {
  return __builtin_amdgcn_wmma_f32_16x16x32_f16(false, a, false, b, (short)0, c,
                                                false, false);
}

// A fragment: 16x32 f16 row-major in LDS (this code's contiguous k order, same as B pack): lane row = lane&15,
// half 0 holds K 0..15, half 1 holds K 16..31 -> 2x ds b128.
__device__ __forceinline__ v16h load_a(const _Float16* base, int ld, int k0, int lane) {
  int row = lane & 15, hf = (lane >> 4) & 1;
  const _Float16* p0 = base + row * ld + k0 + 16 * hf;
  union { v16h v; v8h h[2]; } u;
  u.h[0] = *(const v8h*)(p0);
  u.h[1] = *(const v8h*)(p0 + 8);
  return u.v;
}

// B fragment from pre-swizzled weights: 2x global b128, fully coalesced.
__device__ __forceinline__ v16h load_bp(const _Float16* Bp, int NT, int kk, int nt, int lane) {
  const _Float16* p = Bp + (((size_t)(kk * NT + nt)) << 9) + lane * 16;
  union { v16h v; v8h h[2]; } u;
  u.h[0] = *(const v8h*)(p);
  u.h[1] = *(const v8h*)(p + 8);
  return u.v;
}

// ---------------- weight prep: swizzle f32 -> f16 fragment layout ----------
__global__ void pack_frag(_Float16* __restrict__ dst, const float* __restrict__ src,
                          int KC, int NT, int Kv, int Nv, int srcld) {
  int i = blockIdx.x * 256 + threadIdx.x;
  if (i >= KC * NT * 512) return;
  int frag = i >> 9, r = i & 511, lane = r >> 4, j = r & 15;
  int kk = frag / NT, nt = frag % NT;
  int k = kk * 32 + 16 * (lane >> 4) + j;
  int n = nt * 16 + (lane & 15);
  dst[i] = (k < Kv && n < Nv) ? (_Float16)src[(size_t)k * srcld + n] : (_Float16)0.f;
}

__global__ void pack_frag_T(_Float16* __restrict__ dst, const float* __restrict__ src,
                            int KC, int NT, int Nv, int S) {
  // logical W[k][n] = src[n*S + k]
  int i = blockIdx.x * 256 + threadIdx.x;
  if (i >= KC * NT * 512) return;
  int frag = i >> 9, r = i & 511, lane = r >> 4, j = r & 15;
  int kk = frag / NT, nt = frag % NT;
  int k = kk * 32 + 16 * (lane >> 4) + j;
  int n = nt * 16 + (lane & 15);
  dst[i] = (n < Nv) ? (_Float16)src[(size_t)n * S + k] : (_Float16)0.f;
}

// ---------------- layer building blocks (M=64 per block) ----------------
// 256->256 layer: each wave owns 2 N-tiles; B fragments reused over 4 M-tiles.
__device__ __forceinline__ void big_layer(
    const _Float16 (*A)[HID], const _Float16* __restrict__ Bp,
    _Float16 (*Hout)[HID], const float* __restrict__ bias, int wave, int lane) {
  int col = lane & 15, hf = (lane >> 4) & 1;
  v8f acc[4][2];
#pragma unroll
  for (int m = 0; m < 4; ++m) { acc[m][0] = {}; acc[m][1] = {}; }
  for (int kk = 0; kk < 8; ++kk) {
    v16h b0 = load_bp(Bp, 16, kk, wave * 2 + 0, lane);
    v16h b1 = load_bp(Bp, 16, kk, wave * 2 + 1, lane);
#pragma unroll
    for (int m = 0; m < 4; ++m) {
      v16h a = load_a(&A[m * 16][0], HID, kk * 32, lane);
      acc[m][0] = wmma16(a, b0, acc[m][0]);
      acc[m][1] = wmma16(a, b1, acc[m][1]);
    }
  }
#pragma unroll
  for (int m = 0; m < 4; ++m)
#pragma unroll
    for (int nt = 0; nt < 2; ++nt) {
      int n0 = (wave * 2 + nt) * 16;
      float bb = bias[n0 + col];
#pragma unroll
      for (int r = 0; r < 8; ++r)
        Hout[m * 16 + r + 8 * hf][n0 + col] =
            (_Float16)fmaxf(acc[m][nt][r] + bb, 0.f);
    }
}

// 256->256 backward layer: output masked by relu mask of Hmask, no bias.
__device__ __forceinline__ void big_layer_masked(
    const _Float16 (*A)[HID], const _Float16* __restrict__ Bp,
    const _Float16 (*Hmask)[HID], _Float16 (*Gout)[HID], int wave, int lane) {
  int col = lane & 15, hf = (lane >> 4) & 1;
  v8f acc[4][2];
#pragma unroll
  for (int m = 0; m < 4; ++m) { acc[m][0] = {}; acc[m][1] = {}; }
  for (int kk = 0; kk < 8; ++kk) {
    v16h b0 = load_bp(Bp, 16, kk, wave * 2 + 0, lane);
    v16h b1 = load_bp(Bp, 16, kk, wave * 2 + 1, lane);
#pragma unroll
    for (int m = 0; m < 4; ++m) {
      v16h a = load_a(&A[m * 16][0], HID, kk * 32, lane);
      acc[m][0] = wmma16(a, b0, acc[m][0]);
      acc[m][1] = wmma16(a, b1, acc[m][1]);
    }
  }
#pragma unroll
  for (int m = 0; m < 4; ++m)
#pragma unroll
    for (int nt = 0; nt < 2; ++nt) {
      int n0 = (wave * 2 + nt) * 16;
#pragma unroll
      for (int r = 0; r < 8; ++r) {
        int row = m * 16 + r + 8 * hf;
        Gout[row][n0 + col] = ((float)Hmask[row][n0 + col] > 0.f)
                                  ? (_Float16)acc[m][nt][r] : (_Float16)0.f;
      }
    }
}

// 32->256 input layer (K=32, one k-chunk).
__device__ __forceinline__ void in_layer(
    const _Float16 (*In)[32], const _Float16* __restrict__ Bp,
    _Float16 (*Hout)[HID], const float* __restrict__ bias, int wave, int lane) {
  int col = lane & 15, hf = (lane >> 4) & 1;
  v16h b0 = load_bp(Bp, 16, 0, wave * 2 + 0, lane);
  v16h b1 = load_bp(Bp, 16, 0, wave * 2 + 1, lane);
#pragma unroll
  for (int m = 0; m < 4; ++m) {
    v16h a = load_a(&In[m * 16][0], 32, 0, lane);
    v8f c0 = {}, c1 = {};
    c0 = wmma16(a, b0, c0);
    c1 = wmma16(a, b1, c1);
#pragma unroll
    for (int nt = 0; nt < 2; ++nt) {
      int n0 = (wave * 2 + nt) * 16;
      float bb = bias[n0 + col];
#pragma unroll
      for (int r = 0; r < 8; ++r) {
        float v = (nt ? c1[r] : c0[r]) + bb;
        Hout[m * 16 + r + 8 * hf][n0 + col] = (_Float16)fmaxf(v, 0.f);
      }
    }
  }
}

// 256->16 narrow layer: waves 0..3, one M-tile each.
__device__ __forceinline__ v8f nar_layer(const _Float16 (*A)[HID],
                                         const _Float16* __restrict__ Bp,
                                         int m, int lane) {
  v8f c = {};
  for (int kk = 0; kk < 8; ++kk) {
    v16h a = load_a(&A[m * 16][0], HID, kk * 32, lane);
    v16h b = load_bp(Bp, 1, kk, 0, lane);
    c = wmma16(a, b, c);
  }
  return c;
}

// ---------------- main fused kernel: 64 points per block ----------------
__global__ __launch_bounds__(256) void monosdf_main(
    const float* __restrict__ rays_o, const float* __restrict__ rays_d,
    const float* __restrict__ tnear, const float* __restrict__ tfar,
    const float* __restrict__ grid,
    const _Float16* __restrict__ GW1p, const float* __restrict__ gb1,
    const _Float16* __restrict__ GW2p, const float* __restrict__ gb2,
    const _Float16* __restrict__ GW2Tp,
    const _Float16* __restrict__ GW3p, const float* __restrict__ gb3,
    const _Float16* __restrict__ GW1Tp,
    const _Float16* __restrict__ CW1p, const float* __restrict__ cb1,
    const _Float16* __restrict__ CW2p, const float* __restrict__ cb2,
    const _Float16* __restrict__ CW3p, const float* __restrict__ cb3,
    const float* __restrict__ gw3f,
    float* __restrict__ o_grd, float* __restrict__ ws_sdf, float* __restrict__ ws_rgb)
{
  __shared__ alignas(16) _Float16 sIn[64][32];
  __shared__ alignas(16) _Float16 sH1[64][HID];
  __shared__ alignas(16) _Float16 sH2[64][HID];   // also reused for bwd seed
  __shared__ alignas(16) _Float16 sG1[64][HID];
  __shared__ float sX[64][3], sD[64][3];
  __shared__ float sEmb[64][8];
  __shared__ float sJ[64][3][8];   // d(emb)/d(x,y,z)
  __shared__ float sGeo[64][16];   // geo layer3 out
  __shared__ float sDx[64][16];    // d sdf / d input

  const int t = threadIdx.x;
  const int lane = t & 31;
  const int wave = t >> 5;
  const int col = lane & 15, hf = (lane >> 4) & 1;
  const int gp0 = blockIdx.x * 64;

  // ---- sample positions ----
  if (t < 64) {
    int gp = gp0 + t;
    int r = gp >> 6, s = gp & 63;
    float dt = (tfar[r] - tnear[r]) * (1.0f / 64.0f);
    float tm = tnear[r] + ((float)s + 0.5f) * dt;
#pragma unroll
    for (int c = 0; c < 3; ++c) {
      float d = rays_d[r * 3 + c];
      sX[t][c] = rays_o[r * 3 + c] + tm * d;
      sD[t][c] = d;
    }
  }
  __syncthreads();

  // ---- trilerp + analytic Jacobian: (point p, channel e), 2 per thread ----
  for (int i = t; i < 512; i += 256) {
    int p = i >> 3, e = i & 7;
    float x = sX[p][0], y = sX[p][1], z = sX[p][2];
    bool inb = (fabsf(x) <= 1.f) && (fabsf(y) <= 1.f) && (fabsf(z) <= 1.f);
    const float sc = 0.5f * 127.0f;
    float ux = (x + 1.f) * sc, uy = (y + 1.f) * sc, uz = (z + 1.f) * sc;
    int ix = (int)fminf(fmaxf(floorf(ux), 0.f), 126.f);
    int iy = (int)fminf(fmaxf(floorf(uy), 0.f), 126.f);
    int iz = (int)fminf(fmaxf(floorf(uz), 0.f), 126.f);
    float fx = ux - (float)ix, fy = uy - (float)iy, fz = uz - (float)iz;
    const float* g0 = grid + (((size_t)(ix * 128 + iy) * 128 + iz) * 8 + e);
    float c000 = g0[0],      c001 = g0[8];
    float c010 = g0[1024],   c011 = g0[1032];
    float c100 = g0[131072], c101 = g0[131080];
    float c110 = g0[132096], c111 = g0[132104];
    float c00 = c000 + (c001 - c000) * fz;
    float c01 = c010 + (c011 - c010) * fz;
    float c10 = c100 + (c101 - c100) * fz;
    float c11 = c110 + (c111 - c110) * fz;
    float c0 = c00 + (c01 - c00) * fy;
    float c1 = c10 + (c11 - c10) * fy;
    float val = c0 + (c1 - c0) * fx;
    float dfx = c1 - c0;
    float dfy = (1.f - fx) * (c01 - c00) + fx * (c11 - c10);
    float dfz = (1.f - fx) * ((1.f - fy) * (c001 - c000) + fy * (c011 - c010))
              + fx * ((1.f - fy) * (c101 - c100) + fy * (c111 - c110));
    if (!inb) { val = 0.f; dfx = 0.f; dfy = 0.f; dfz = 0.f; }
    sEmb[p][e] = val;
    sJ[p][0][e] = dfx * sc;
    sJ[p][1][e] = dfy * sc;
    sJ[p][2][e] = dfz * sc;
  }
  __syncthreads();

  // ---- geo input [64][32] f16 (emb8 + xyz3, zero pad) ----
  for (int i = t; i < 64 * 32; i += 256) {
    int p = i >> 5, k = i & 31;
    float v = (k < 8) ? sEmb[p][k] : ((k < 11) ? sX[p][k - 8] : 0.f);
    sIn[p][k] = (_Float16)v;
  }
  __syncthreads();

  // ---- geo L1, L2 ----
  in_layer(sIn, GW1p, sH1, gb1, wave, lane);
  __syncthreads();
  big_layer(sH1, GW2p, sH2, gb2, wave, lane);
  __syncthreads();

  // ---- geo L3 (waves 0..3, one M-tile each) ----
  if (wave < 4) {
    v8f c = nar_layer(sH2, GW3p, wave, lane);
    float bb = (col < 9) ? gb3[col] : 0.f;
#pragma unroll
    for (int r = 0; r < 8; ++r) sGeo[wave * 16 + r + 8 * hf][col] = c[r] + bb;
  }
  __syncthreads();

  // ---- backward seed in place: H2 <- W3[:,0] * relu_mask(H2) ----
  for (int i = t; i < 64 * HID; i += 256) {
    int p = i >> 8, k = i & 255;
    float h = (float)sH2[p][k];
    sH2[p][k] = (h > 0.f) ? (_Float16)gw3f[k * 9] : (_Float16)0.f;
  }
  __syncthreads();

  // ---- backward L2: G1 = (seed @ W2^T) * relu_mask(H1) ----
  big_layer_masked(sH2, GW2Tp, sH1, sG1, wave, lane);
  __syncthreads();

  // ---- backward L1: dXin = G1 @ W1^T ----
  if (wave < 4) {
    v8f c = nar_layer(sG1, GW1Tp, wave, lane);
#pragma unroll
    for (int r = 0; r < 8; ++r) sDx[wave * 16 + r + 8 * hf][col] = c[r];
  }
  __syncthreads();

  // ---- sdf grads (chain trilerp Jacobian), sdf, color input ----
  if (t < 192) {
    int p = t / 3, c = t % 3;
    float g = sDx[p][8 + c];
#pragma unroll
    for (int e = 0; e < 8; ++e) g += sDx[p][e] * sJ[p][c][e];
    o_grd[(size_t)(gp0 + p) * 3 + c] = g;
  }
  if (t < 64) ws_sdf[gp0 + t] = sGeo[t][0];
  for (int i = t; i < 64 * 32; i += 256) {
    int p = i >> 5, k = i & 31;
    float v = (k < 8) ? sGeo[p][k + 1]
            : (k < 11) ? sX[p][k - 8]
            : (k < 14) ? sD[p][k - 11] : 0.f;
    sIn[p][k] = (_Float16)v;
  }
  __syncthreads();

  // ---- color L1, L2 ----
  in_layer(sIn, CW1p, sH1, cb1, wave, lane);
  __syncthreads();
  big_layer(sH1, CW2p, sH2, cb2, wave, lane);
  __syncthreads();

  // ---- color L3 + sigmoid ----
  if (wave < 4) {
    v8f c = nar_layer(sH2, CW3p, wave, lane);
    if (col < 3) {
      float bb = cb3[col];
#pragma unroll
      for (int r = 0; r < 8; ++r) {
        float v = c[r] + bb;
        ws_rgb[(size_t)(gp0 + wave * 16 + r + 8 * hf) * 3 + col] =
            1.f / (1.f + expf(-v));
      }
    }
  }
}

// ---------------- per-ray volumetric compositing ----------------
__global__ void monosdf_composite(
    const float* __restrict__ tnear, const float* __restrict__ tfar,
    const float* __restrict__ rdn, const float* __restrict__ beta,
    const float* __restrict__ ws_sdf, const float* __restrict__ ws_rgb,
    const float* __restrict__ grd,
    float* __restrict__ o_rgb, float* __restrict__ o_dep,
    float* __restrict__ o_nrm, float* __restrict__ o_acc)
{
  int r = blockIdx.x * blockDim.x + threadIdx.x;
  if (r >= 4096) return;
  float bc = fmaxf(beta[0], 1e-4f);
  float inv2b = 0.5f / bc;
  float dt = (tfar[r] - tnear[r]) * (1.f / 64.f);
  float T = 1.f, acc = 0.f, dep = 0.f;
  float cr = 0.f, cg = 0.f, cb = 0.f, nx = 0.f, ny = 0.f, nz = 0.f;
  for (int s = 0; s < 64; ++s) {
    int gp = r * 64 + s;
    float sdf = ws_sdf[gp];
    float sgn = (sdf > 0.f) ? 1.f : ((sdf < 0.f) ? -1.f : 0.f);
    float sig = inv2b * (1.f + sgn * expm1f(-fabsf(sdf) / bc));
    float e = expf(-sig * dt);
    float w = T * (1.f - e);
    T *= e;
    float tm = tnear[r] + ((float)s + 0.5f) * dt;
    cr += w * ws_rgb[gp * 3 + 0];
    cg += w * ws_rgb[gp * 3 + 1];
    cb += w * ws_rgb[gp * 3 + 2];
    float gx = grd[gp * 3 + 0], gy = grd[gp * 3 + 1], gz = grd[gp * 3 + 2];
    float nn = fmaxf(sqrtf(gx * gx + gy * gy + gz * gz), 1e-12f);
    nx += w * gx / nn; ny += w * gy / nn; nz += w * gz / nn;
    dep += w * tm;
    acc += w;
  }
  o_rgb[r * 3 + 0] = cr; o_rgb[r * 3 + 1] = cg; o_rgb[r * 3 + 2] = cb;
  o_dep[r] = dep / rdn[r];
  o_nrm[r * 3 + 0] = nx; o_nrm[r * 3 + 1] = ny; o_nrm[r * 3 + 2] = nz;
  o_acc[r] = acc;
}

extern "C" void kernel_launch(void* const* d_in, const int* in_sizes, int n_in,
                              void* d_out, int out_size, void* d_ws, size_t ws_size,
                              hipStream_t stream) {
  const float* rays_o = (const float*)d_in[0];
  const float* rays_d = (const float*)d_in[1];
  const float* rdn    = (const float*)d_in[2];
  const float* nearp  = (const float*)d_in[3];
  const float* farp   = (const float*)d_in[4];
  const float* grid   = (const float*)d_in[5];
  const float* gw1 = (const float*)d_in[6];  const float* gb1 = (const float*)d_in[7];
  const float* gw2 = (const float*)d_in[8];  const float* gb2 = (const float*)d_in[9];
  const float* gw3 = (const float*)d_in[10]; const float* gb3 = (const float*)d_in[11];
  const float* cw1 = (const float*)d_in[12]; const float* cb1 = (const float*)d_in[13];
  const float* cw2 = (const float*)d_in[14]; const float* cb2 = (const float*)d_in[15];
  const float* cw3 = (const float*)d_in[16]; const float* cb3 = (const float*)d_in[17];
  const float* beta = (const float*)d_in[18];

  char* ws = (char*)d_ws;
  _Float16* GW1p  = (_Float16*)(ws + OFF_GW1);
  _Float16* GW2p  = (_Float16*)(ws + OFF_GW2);
  _Float16* GW2Tp = (_Float16*)(ws + OFF_GW2T);
  _Float16* GW3p  = (_Float16*)(ws + OFF_GW3);
  _Float16* GW1Tp = (_Float16*)(ws + OFF_GW1T);
  _Float16* CW1p  = (_Float16*)(ws + OFF_CW1);
  _Float16* CW2p  = (_Float16*)(ws + OFF_CW2);
  _Float16* CW3p  = (_Float16*)(ws + OFF_CW3);
  float* ws_sdf = (float*)(ws + OFF_SDF);
  float* ws_rgb = (float*)(ws + OFF_RGB);

  float* out   = (float*)d_out;
  float* o_rgb = out;            // 4096*3
  float* o_dep = out + 12288;    // 4096
  float* o_nrm = out + 16384;    // 4096*3
  float* o_acc = out + 28672;    // 4096
  float* o_grd = out + 32768;    // 262144*3

  // weight swizzle into fragment layout
  pack_frag  <<<32,  256, 0, stream>>>(GW1p,  gw1, 1, 16, 11, 256, 256);
  pack_frag  <<<256, 256, 0, stream>>>(GW2p,  gw2, 8, 16, 256, 256, 256);
  pack_frag_T<<<256, 256, 0, stream>>>(GW2Tp, gw2, 8, 16, 256, 256);
  pack_frag  <<<16,  256, 0, stream>>>(GW3p,  gw3, 8, 1, 256, 9, 9);
  pack_frag_T<<<16,  256, 0, stream>>>(GW1Tp, gw1, 8, 1, 11, 256);
  pack_frag  <<<32,  256, 0, stream>>>(CW1p,  cw1, 1, 16, 14, 256, 256);
  pack_frag  <<<256, 256, 0, stream>>>(CW2p,  cw2, 8, 16, 256, 256, 256);
  pack_frag  <<<16,  256, 0, stream>>>(CW3p,  cw3, 8, 1, 256, 3, 3);

  monosdf_main<<<4096, 256, 0, stream>>>(
      rays_o, rays_d, nearp, farp, grid,
      GW1p, gb1, GW2p, gb2, GW2Tp, GW3p, gb3, GW1Tp,
      CW1p, cb1, CW2p, cb2, CW3p, cb3, gw3,
      o_grd, ws_sdf, ws_rgb);

  monosdf_composite<<<16, 256, 0, stream>>>(
      nearp, farp, rdn, beta, ws_sdf, ws_rgb, o_grd,
      o_rgb, o_dep, o_nrm, o_acc);
}
